// Speller_56916906606655
// MI455X (gfx1250) — matrix operand order repaired and checked by reading, not compiled, not run on hardware
//
#include <hip/hip_runtime.h>

typedef unsigned short u16;
typedef unsigned int   u32;
typedef __attribute__((ext_vector_type(16))) __bf16 v16bf;
typedef __attribute__((ext_vector_type(8)))  float  v8f;
typedef __attribute__((ext_vector_type(4)))  float  v4f;
typedef __attribute__((ext_vector_type(4)))  u32    v4u;
typedef __attribute__((ext_vector_type(4)))  int    v4i;
typedef __attribute__((address_space(1)))    v4i    as1_v4i;
typedef __attribute__((address_space(3)))    v4i    as3_v4i;

struct Frag32B { v4u lo, hi; };   // 32 bytes == v16bf

static __device__ __forceinline__ u16 f32_to_bf16(float f) {
  u32 u = __float_as_uint(f);
  u32 r = u + 0x7FFFu + ((u >> 16) & 1u);   // round-to-nearest-even
  return (u16)(r >> 16);
}
static __device__ __forceinline__ u32 pack_bf16x2(float a, float b) {
  return (u32)f32_to_bf16(a) | ((u32)f32_to_bf16(b) << 16);
}
static __device__ __forceinline__ float bf16_to_f32(u16 h) {
  return __uint_as_float(((u32)h) << 16);
}
static __device__ __forceinline__ float sigm(float x) {
  return 1.0f / (1.0f + __expf(-x));
}

#if defined(__has_builtin)
#if __has_builtin(__builtin_amdgcn_global_load_async_to_lds_b128)
#define HAVE_ASYNC_LDS 1
#endif
#endif

// ---------------------------------------------------------------------------
// bf16 WMMA GEMM: out[M,N] = A[M,K] @ W[N,K]^T + bias[N]
// NB = 16x16 output tiles per wave (register blocking along N; A frag reused).
// Block: 256 threads = 8 waves -> block tile = 16(M) x (8*NB*16)(N).
// grid.x = M/16, grid.y = N/(128*NB). Dynamic LDS = 16*K*2 bytes. K % 128 == 0.
// A staged into LDS: async b128 copies when already bf16, convert path for f32.
// ---------------------------------------------------------------------------
template <int NB>
__global__ void gemm_wmma_kernel(const void* __restrict__ Av, int a_is_f32, int lda,
                                 const u16* __restrict__ W, const float* __restrict__ bias,
                                 float* __restrict__ outf, u16* __restrict__ outh,
                                 int ldo, int K) {
  extern __shared__ char smem[];
  u16* sA = (u16*)smem;
  const int tid  = threadIdx.x;
  const int wave = tid >> 5;
  const int lane = tid & 31;
  const int m0 = blockIdx.x * 16;
  const int n0 = (blockIdx.y * 8 + wave) * (16 * NB);

  // ---- stage A tile [16,K] bf16 into LDS (row = tid>>4, 16B chunks) ----
  const int r  = tid >> 4;
  const int c0 = (tid & 15) * 8;
  if (a_is_f32) {
    const float* ar = (const float*)Av + (size_t)(m0 + r) * lda;
    for (int c = c0; c < K; c += 128) {
      v4f f0 = *(const v4f*)(ar + c);
      v4f f1 = *(const v4f*)(ar + c + 4);
      v4u pk;
      pk[0] = pack_bf16x2(f0[0], f0[1]);
      pk[1] = pack_bf16x2(f0[2], f0[3]);
      pk[2] = pack_bf16x2(f1[0], f1[1]);
      pk[3] = pack_bf16x2(f1[2], f1[3]);
      *(v4u*)(sA + r * K + c) = pk;
    }
  } else {
    const u16* ag = (const u16*)Av + (size_t)(m0 + r) * lda;
#ifdef HAVE_ASYNC_LDS
    for (int c = c0; c < K; c += 128) {
      __builtin_amdgcn_global_load_async_to_lds_b128(
          (as1_v4i*)(ag + c),
          (as3_v4i*)(sA + r * K + c),
          0, 0);
    }
#if __has_builtin(__builtin_amdgcn_s_wait_asynccnt)
    __builtin_amdgcn_s_wait_asynccnt(0);
#else
    asm volatile("s_wait_asynccnt 0x0" ::: "memory");
#endif
#else
    for (int c = c0; c < K; c += 128)
      *(v4u*)(sA + r * K + c) = *(const v4u*)(ag + c);
#endif
  }
  __syncthreads();

  // ---- WMMA main loop ----
  const int mlane = lane & 15;
  const int hi    = lane >> 4;
  v8f acc[NB] = {};
  const u16* __restrict__ arow  = sA + mlane * K;
  const u16* __restrict__ wbase = W + (size_t)(n0 + mlane) * K;

  for (int k0 = 0; k0 < K; k0 += 32) {
    // A 16x32 bf16 fragment: lanes 0-15 M=lane, v0-3 K=hi*8+0..7, v4-7 K=16+hi*8+0..7
    Frag32B fa;
    const u16* ap = arow + k0 + hi * 8;
    fa.lo = *(const v4u*)(ap);
    fa.hi = *(const v4u*)(ap + 16);
    v16bf a = __builtin_bit_cast(v16bf, fa);
#pragma unroll
    for (int j = 0; j < NB; ++j) {
      // B 32x16 bf16 fragment: lane N = n0+j*16+(lane&15), v0-3 K=hi*8+0..7, v4-7 K=16+hi*8+0..7
      const u16* bp = wbase + (size_t)j * 16 * K + k0 + hi * 8;
      Frag32B fb;
      fb.lo = *(const v4u*)(bp);
      fb.hi = *(const v4u*)(bp + 16);
      v16bf b = __builtin_bit_cast(v16bf, fb);
      acc[j] = __builtin_amdgcn_wmma_f32_16x16x32_bf16(false, a, false, b,
                                                       (short)0, acc[j], false, false);
    }
  }

  // ---- epilogue ----
#pragma unroll
  for (int j = 0; j < NB; ++j) {
    const int n = n0 + j * 16 + mlane;
    const float bvv = bias ? bias[n] : 0.0f;
#pragma unroll
    for (int rr = 0; rr < 8; ++rr) {
      const int m = m0 + rr + hi * 8;   // C/D layout: lanes 16-31 hold M = rr+8
      float v = acc[j][rr] + bvv;
      if (outf) outf[(size_t)m * ldo + n] = v;
      else      outh[(size_t)m * ldo + n] = f32_to_bf16(v);
    }
  }
}

// ---------------------------------------------------------------------------
// Weight prep: concat + f32->bf16 convert, bias sums. Flat segmented index.
// ---------------------------------------------------------------------------
__global__ void prep_weights_kernel(
    const float* __restrict__ Wih1, const float* __restrict__ Whh1,
    const float* __restrict__ Wih2, const float* __restrict__ Whh2,
    const float* __restrict__ Wq,   const float* __restrict__ Wk,
    const float* __restrict__ Wv,   const float* __restrict__ emb,
    const float* __restrict__ bih1, const float* __restrict__ bhh1,
    const float* __restrict__ bih2, const float* __restrict__ bhh2,
    u16* __restrict__ W1, u16* __restrict__ W2, u16* __restrict__ Wqh,
    u16* __restrict__ Wkh, u16* __restrict__ Wvh, u16* __restrict__ embh,
    float* __restrict__ b1, float* __restrict__ b2) {
  long i = (long)blockIdx.x * 256 + threadIdx.x;
  const long S0 = 2048L * 896;
  const long S1 = S0 + 512L * 640;
  const long S2 = S1 + 128L * 128;
  const long S3 = S2 + 128L * 512;
  const long S4 = S3 + 128L * 512;
  const long S5 = S4 + 30L * 256;
  const long S6 = S5 + 2048;
  const long S7 = S6 + 512;
  if (i < S0) {
    long n = i / 896, k = i - n * 896;
    float v = (k < 384) ? Wih1[n * 384 + k] : Whh1[n * 512 + (k - 384)];
    W1[i] = f32_to_bf16(v);
  } else if (i < S1) {
    long j = i - S0, n = j / 640, k = j - n * 640;
    float v = (k < 512) ? Wih2[n * 512 + k] : Whh2[n * 128 + (k - 512)];
    W2[j] = f32_to_bf16(v);
  } else if (i < S2) { long j = i - S1; Wqh[j]  = f32_to_bf16(Wq[j]); }
  else   if (i < S3) { long j = i - S2; Wkh[j]  = f32_to_bf16(Wk[j]); }
  else   if (i < S4) { long j = i - S3; Wvh[j]  = f32_to_bf16(Wv[j]); }
  else   if (i < S5) { long j = i - S4; embh[j] = f32_to_bf16(emb[j]); }
  else   if (i < S6) { long j = i - S5; b1[j] = bih1[j] + bhh1[j]; }
  else   if (i < S7) { long j = i - S6; b2[j] = bih2[j] + bhh2[j]; }
}

__global__ void zero_kernel(u32* __restrict__ p, long n) {
  long i = (long)blockIdx.x * 256 + threadIdx.x;
  if (i < n) p[i] = 0u;
}

// context0 = mean_s values[b,s,:]  -> ctxf (f32) and xcat1[:,0:128] (bf16)
__global__ void context0_kernel(const u16* __restrict__ values,
                                float* __restrict__ ctxf, u16* __restrict__ xcat1) {
  int i = blockIdx.x * 256 + threadIdx.x;     // 128*128
  int b = i >> 7, p = i & 127;
  const u16* vp = values + (size_t)b * 400 * 128 + p;
  float acc = 0.0f;
  for (int s = 0; s < 400; ++s) acc += bf16_to_f32(vp[(size_t)s * 128]);
  acc *= (1.0f / 400.0f);
  ctxf[i] = acc;
  xcat1[b * 896 + p] = f32_to_bf16(acc);
}

// gather step-t char embeddings into xcat1[:,128:384]
__global__ void gather_x_kernel(const int* __restrict__ y, const u16* __restrict__ embh,
                                u16* __restrict__ xcat1, int t) {
  int i = blockIdx.x * 256 + threadIdx.x;     // 128*256
  int b = i >> 8, e = i & 255;
  int tok = (t == 0) ? 0 : y[b * 300 + t];    // SOS == 0
  xcat1[b * 896 + 128 + e] = embh[tok * 256 + e];
}

// LSTM1 activation: gates [128,2048] (i,f,g,o) -> c1, h1(bf16 into xcat1/xcat2)
__global__ void act1_kernel(const float* __restrict__ g, float* __restrict__ c1,
                            u16* __restrict__ xcat1, u16* __restrict__ xcat2) {
  int i = blockIdx.x * 256 + threadIdx.x;     // 128*512
  int b = i >> 9, j = i & 511;
  const float* gb = g + (size_t)b * 2048;
  float ig = sigm(gb[j]);
  float fg = sigm(gb[512 + j]);
  float gg = tanhf(gb[1024 + j]);
  float og = sigm(gb[1536 + j]);
  float c = fg * c1[i] + ig * gg;
  c1[i] = c;
  float h = og * tanhf(c);
  u16 hb = f32_to_bf16(h);
  xcat1[b * 896 + 384 + j] = hb;
  xcat2[b * 640 + j] = hb;
}

// LSTM2 activation: gates [128,512] -> c2, h2 (f32 + bf16 into xcat2[:,512:640])
__global__ void act2_kernel(const float* __restrict__ g, float* __restrict__ c2,
                            float* __restrict__ h2f, u16* __restrict__ xcat2) {
  int i = blockIdx.x * 256 + threadIdx.x;     // 128*128
  int b = i >> 7, j = i & 127;
  const float* gb = g + (size_t)b * 512;
  float ig = sigm(gb[j]);
  float fg = sigm(gb[128 + j]);
  float gg = tanhf(gb[256 + j]);
  float og = sigm(gb[384 + j]);
  float c = fg * c2[i] + ig * gg;
  c2[i] = c;
  float h = og * tanhf(c);
  h2f[i] = h;
  xcat2[b * 640 + 512 + j] = f32_to_bf16(h);
}

// masked softmax attention; one block per batch element
__global__ void attention_kernel(const u16* __restrict__ keys, const u16* __restrict__ values,
                                 const float* __restrict__ qf, const int* __restrict__ lens,
                                 float* __restrict__ ctxf, u16* __restrict__ xcat1,
                                 float* __restrict__ attn_out) {
  __shared__ float sq[128];
  __shared__ float sw[400];
  __shared__ float sred[256];
  const int b = blockIdx.x, tid = threadIdx.x;
  if (tid < 128) sq[tid] = qf[b * 128 + tid];
  __syncthreads();
  const int len = lens[b];
  const float scale = 0.08838834764831845f;   // 1/sqrt(128)
  for (int s = tid; s < 400; s += 256) {
    const u16* kp = keys + ((size_t)(b * 400 + s)) * 128;
    float acc = 0.0f;
    for (int p = 0; p < 128; p += 8) {
      v4u kv4 = *(const v4u*)(kp + p);
#pragma unroll
      for (int j = 0; j < 4; ++j) {
        u32 w2 = kv4[j];
        acc += __uint_as_float(w2 << 16)         * sq[p + 2 * j];
        acc += __uint_as_float(w2 & 0xFFFF0000u) * sq[p + 2 * j + 1];
      }
    }
    sw[s] = (s < len) ? acc * scale : -1e9f;
  }
  __syncthreads();
  // max reduce
  float m = -3.4e38f;
  for (int s = tid; s < 400; s += 256) m = fmaxf(m, sw[s]);
  sred[tid] = m; __syncthreads();
  for (int off = 128; off > 0; off >>= 1) {
    if (tid < off) sred[tid] = fmaxf(sred[tid], sred[tid + off]);
    __syncthreads();
  }
  const float mx = sred[0];
  __syncthreads();
  // exp + sum reduce
  float ssum = 0.0f;
  for (int s = tid; s < 400; s += 256) { float e = __expf(sw[s] - mx); sw[s] = e; ssum += e; }
  sred[tid] = ssum; __syncthreads();
  for (int off = 128; off > 0; off >>= 1) {
    if (tid < off) sred[tid] = sred[tid] + sred[tid + off];
    __syncthreads();
  }
  const float inv = 1.0f / sred[0];
  __syncthreads();
  for (int s = tid; s < 400; s += 256) sw[s] *= inv;
  __syncthreads();
  // context = w @ values
  if (tid < 128) {
    const int p = tid;
    const u16* vp = values + (size_t)b * 400 * 128 + p;
    float acc = 0.0f;
    for (int s = 0; s < 400; ++s) acc += bf16_to_f32(vp[(size_t)s * 128]) * sw[s];
    ctxf[b * 128 + p] = acc;
    xcat1[b * 896 + p] = f32_to_bf16(acc);
  }
  if (b == 0) {
    for (int s = tid; s < 400; s += 256) attn_out[s] = sw[s];
  }
}

// logits = [context|h2] @ embedding^T + b_out, written to predictions[b,v,t]
__global__ void logits_kernel(const float* __restrict__ ctx, const float* __restrict__ h2,
                              const float* __restrict__ emb, const float* __restrict__ bout,
                              float* __restrict__ out, int t) {
  int i = blockIdx.x * 256 + threadIdx.x;     // 128*30
  if (i >= 128 * 30) return;
  int b = i / 30, v = i - b * 30;
  const float* er = emb + (size_t)v * 256;
  const float* cb = ctx + (size_t)b * 128;
  const float* hb = h2  + (size_t)b * 128;
  float acc = bout[v];
  for (int e = 0; e < 128; ++e) acc += cb[e] * er[e];
  for (int e = 0; e < 128; ++e) acc += hb[e] * er[128 + e];
  out[(size_t)b * 9000 + (size_t)v * 300 + t] = acc;   // 9000 = V*T
}

// ---------------------------------------------------------------------------
extern "C" void kernel_launch(void* const* d_in, const int* in_sizes, int n_in,
                              void* d_out, int out_size, void* d_ws, size_t ws_size,
                              hipStream_t stream) {
  const float* enc  = (const float*)d_in[0];
  const int*   lens = (const int*)d_in[1];
  const int*   y    = (const int*)d_in[2];
  const float* emb  = (const float*)d_in[3];
  const float* Wih1 = (const float*)d_in[4];
  const float* Whh1 = (const float*)d_in[5];
  const float* bih1 = (const float*)d_in[6];
  const float* bhh1 = (const float*)d_in[7];
  const float* Wih2 = (const float*)d_in[8];
  const float* Whh2 = (const float*)d_in[9];
  const float* bih2 = (const float*)d_in[10];
  const float* bhh2 = (const float*)d_in[11];
  const float* Wk   = (const float*)d_in[12];
  const float* bk   = (const float*)d_in[13];
  const float* Wv   = (const float*)d_in[14];
  const float* bv   = (const float*)d_in[15];
  const float* Wq   = (const float*)d_in[16];
  const float* bq   = (const float*)d_in[17];
  const float* bout = (const float*)d_in[18];
  float* out = (float*)d_out;
  (void)in_sizes; (void)n_in; (void)out_size; (void)ws_size;

  char* w = (char*)d_ws;
  auto carve = [&](size_t bytes) { char* p = w; w += (bytes + 255) & ~(size_t)255; return p; };
  u16*  W1h   = (u16*)carve(2048UL * 896 * 2);
  u16*  W2h   = (u16*)carve(512UL * 640 * 2);
  u16*  Wqh   = (u16*)carve(128UL * 128 * 2);
  u16*  Wkh   = (u16*)carve(128UL * 512 * 2);
  u16*  Wvh   = (u16*)carve(128UL * 512 * 2);
  u16*  embh  = (u16*)carve(30UL * 256 * 2);
  float* b1   = (float*)carve(2048UL * 4);
  float* b2   = (float*)carve(512UL * 4);
  u16*  keysh = (u16*)carve(128UL * 400 * 128 * 2);
  u16*  valsh = (u16*)carve(128UL * 400 * 128 * 2);
  // contiguous zero-init state block (all sizes multiples of 256B):
  float* c1   = (float*)carve(128UL * 512 * 4);
  float* c2   = (float*)carve(128UL * 128 * 4);
  float* h2f  = (float*)carve(128UL * 128 * 4);
  float* ctxf = (float*)carve(128UL * 128 * 4);
  u16*  xcat1 = (u16*)carve(128UL * 896 * 2);   // [context|char_embed|h1] bf16
  u16*  xcat2 = (u16*)carve(128UL * 640 * 2);   // [h1|h2] bf16
  float* g1   = (float*)carve(128UL * 2048 * 4);
  float* g2   = (float*)carve(128UL * 512 * 4);
  float* qf   = (float*)carve(128UL * 128 * 4);

  // ---- one-shot precompute ----
  prep_weights_kernel<<<9064, 256, 0, stream>>>(Wih1, Whh1, Wih2, Whh2, Wq, Wk, Wv, emb,
                                                bih1, bhh1, bih2, bhh2,
                                                W1h, W2h, Wqh, Wkh, Wvh, embh, b1, b2);
  // keys/values: [51200,512] x [512,128]^T, bf16 output (NB=1 -> 8 waves cover N=128)
  gemm_wmma_kernel<1><<<dim3(3200, 1), 256, 16 * 512 * 2, stream>>>(
      enc, 1, 512, Wkh, bk, nullptr, keysh, 128, 512);
  gemm_wmma_kernel<1><<<dim3(3200, 1), 256, 16 * 512 * 2, stream>>>(
      enc, 1, 512, Wvh, bv, nullptr, valsh, 128, 512);
  // zero state block: 851,968 bytes = 212,992 dwords
  zero_kernel<<<832, 256, 0, stream>>>((u32*)c1, 212992L);
  context0_kernel<<<64, 256, 0, stream>>>(valsh, ctxf, xcat1);

  const size_t PRED = 128UL * 30 * 300;
  for (int t = 0; t < 300; ++t) {
    gather_x_kernel<<<128, 256, 0, stream>>>(y, embh, xcat1, t);
    // LSTM1 gates: [128,896] x [896,2048]^T  (NB=4: block tile 16x512)
    gemm_wmma_kernel<4><<<dim3(8, 4), 256, 16 * 896 * 2, stream>>>(
        xcat1, 0, 896, W1h, b1, g1, nullptr, 2048, 896);
    act1_kernel<<<256, 256, 0, stream>>>(g1, c1, xcat1, xcat2);
    // LSTM2 gates: [128,640] x [640,512]^T  (NB=4: one block col covers N=512)
    gemm_wmma_kernel<4><<<dim3(8, 1), 256, 16 * 640 * 2, stream>>>(
        xcat2, 0, 640, W2h, b2, g2, nullptr, 512, 640);
    act2_kernel<<<64, 256, 0, stream>>>(g2, c2, h2f, xcat2);
    // q = h2 @ Wq^T + bq  (A = h2 slice of xcat2, lda=640; NB=1 covers N=128)
    gemm_wmma_kernel<1><<<dim3(8, 1), 256, 16 * 128 * 2, stream>>>(
        xcat2 + 512, 0, 640, Wqh, bq, qf, nullptr, 128, 128);
    attention_kernel<<<128, 256, 0, stream>>>(keysh, valsh, qf, lens, ctxf, xcat1,
                                              out + PRED + (size_t)t * 400);
    logits_kernel<<<15, 256, 0, stream>>>(ctxf, h2f, emb, bout, out, t);
  }
}
